// RBFLayer_44023414784041
// MI455X (gfx1250) — matrix operand order repaired and checked by reading, not compiled, not run on hardware
//
#include <hip/hip_runtime.h>
#include <math.h>

// ---------------------------------------------------------------------------
// RBF distance layer for MI455X (gfx1250, wave32, WMMA).
// dist[b,k] = sqrt(max(||x_b||^2 + ||c_k||^2 - 2 x_b.c_k, 1e-12))
//
// Numerics: norms are exact fp32; cross term uses one bf16 WMMA pass.
// For N(0,1) data at D=3072, dist^2 ~ 6144 while cross ~ +-55, and bf16-RNE
// product noise is sigma ~ 0.05 on cross -> ~1e-5 relative error on dist
// (<= 5e-5 worst case over all 2^21 outputs). No small distances exist
// (min dist^2 ~ 5600), so there is no cancellation regime.
//
// 128x128x32 block tile, double-buffered LDS (40 KB), 8 waves (4x2),
// each wave 32x64 = 2x4 WMMA tiles -> 8 v_wmma_f32_16x16x32_bf16 per stage.
// ---------------------------------------------------------------------------

typedef __attribute__((ext_vector_type(16))) __bf16 v16bf;
typedef __attribute__((ext_vector_type(8)))  __bf16 v8bf;
typedef __attribute__((ext_vector_type(2)))  __bf16 v2bf;
typedef __attribute__((ext_vector_type(8)))  float  v8f;

namespace {
constexpr int kB  = 8192;   // samples
constexpr int kK  = 256;    // centers
constexpr int kD  = 3072;   // feature dim
constexpr int BM  = 128;    // block tile M
constexpr int BN  = 128;    // block tile N
constexpr int BK  = 32;     // k-step (one bf16 WMMA K)
constexpr int NKI = kD / BK;        // 96 k-steps
constexpr int STR = BK + 8;         // LDS row stride in halves (conflict-free pad)
constexpr float kEps = 1e-12f;
}

// ---------------- kernel 1: squared row norms of inputs & centers -----------
__global__ __launch_bounds__(256) void rbf_sqnorm(const float* __restrict__ x,
                                                  const float* __restrict__ c,
                                                  float* __restrict__ sq) {
  const int row = blockIdx.x;
  const float* p = (row < kB) ? (x + (size_t)row * kD)
                              : (c + (size_t)(row - kB) * kD);
  const float4* p4 = (const float4*)p;
  float s = 0.0f;
#pragma unroll
  for (int i = 0; i < 3; ++i) {            // 3 * 256 * 4 = 3072 elements
    float4 v = p4[threadIdx.x + i * 256];
    s += v.x * v.x + v.y * v.y + v.z * v.z + v.w * v.w;
  }
  __shared__ float red[256];
  red[threadIdx.x] = s;
  __syncthreads();
#pragma unroll
  for (int off = 128; off > 0; off >>= 1) {
    if ((int)threadIdx.x < off) red[threadIdx.x] += red[threadIdx.x + off];
    __syncthreads();
  }
  if (threadIdx.x == 0) sq[row] = red[0];
}

// pack two fp32 -> one dword of two bf16 (RNE), v_cvt_pk_bf16_f32
__device__ __forceinline__ unsigned pk2(float a, float b) {
#if __has_builtin(__builtin_amdgcn_cvt_pk_bf16_f32)
  auto v = __builtin_amdgcn_cvt_pk_bf16_f32(a, b);
  return __builtin_bit_cast(unsigned, v);
#else
  v2bf v = {(__bf16)a, (__bf16)b};
  return __builtin_bit_cast(unsigned, v);
#endif
}

// ---------------- kernel 2: WMMA GEMM + fused distance epilogue -------------
__global__ __launch_bounds__(256) void rbf_gemm_wmma(const float* __restrict__ x,
                                                     const float* __restrict__ c,
                                                     const float* __restrict__ sq,
                                                     float* __restrict__ out) {
  // [buf][row][k], padded row stride (80B) -> conflict-free b128 frag loads
  __shared__ __align__(16) __bf16 sA[2][BM][STR];   // 20 KB
  __shared__ __align__(16) __bf16 sB[2][BN][STR];   // 20 KB

  const int t    = threadIdx.x;
  const int lane = t & 31;
  const int wave = t >> 5;     // 8 waves
  const int wm   = wave & 3;   // 4 wave-rows -> 32 M each
  const int wn   = wave >> 2;  // 2 wave-cols -> 64 N each
  const int hw   = lane >> 4;  // half-wave
  const int lr   = lane & 15;

  const int m0 = blockIdx.y * BM;
  const int n0 = blockIdx.x * BN;

  const int qrow = t >> 3;           // staging row
  const int qcol = (t & 7) << 2;     // staging col (floats / halves)

  // divergent 32-bit element offsets; uniform base advances per k-step so the
  // compiler can use saddr-form global_load_b128 with immediate offsets.
  const int aOff = (m0 + qrow) * kD + qcol;
  const int bOff = (n0 + qrow) * kD + qcol;

  float4 aReg[4];   // 128x32 tile: 1024 float4 / 256 threads
  float4 bReg[4];

  auto globalLoad = [&](int kt) {
    const float* xa = x + (size_t)kt * BK;   // uniform
    const float* cb = c + (size_t)kt * BK;   // uniform
#pragma unroll
    for (int i = 0; i < 4; ++i) {
      aReg[i] = *(const float4*)(xa + (aOff + i * (32 * kD)));
      bReg[i] = *(const float4*)(cb + (bOff + i * (32 * kD)));
    }
  };

  auto ldsStore = [&](int buf) {
#pragma unroll
    for (int i = 0; i < 4; ++i) {
      const int row = qrow + i * 32;
      *(uint2*)&sA[buf][row][qcol] =
          make_uint2(pk2(aReg[i].x, aReg[i].y), pk2(aReg[i].z, aReg[i].w));
      *(uint2*)&sB[buf][row][qcol] =
          make_uint2(pk2(bReg[i].x, bReg[i].y), pk2(bReg[i].z, bReg[i].w));
    }
  };

  // A frag (16x32): lane<16 holds row M=lr, K 0-7 & 16-23; lane>=16: K 8-15 & 24-31
  auto fragA = [&](int buf, int mt) -> v16bf {
    const __bf16* p = &sA[buf][wm * 32 + mt * 16 + lr][hw ? 8 : 0];
    const v8bf q0 = *(const v8bf*)p;
    const v8bf q1 = *(const v8bf*)(p + 16);
    return __builtin_shufflevector(q0, q1, 0, 1, 2, 3, 4, 5, 6, 7,
                                   8, 9, 10, 11, 12, 13, 14, 15);
  };
  // B frag (32x16): lane holds column N=lr; lane<16: K 0-7 & 16-23, lane>=16: K 8-15 & 24-31
  auto fragB = [&](int buf, int nt) -> v16bf {
    const __bf16* p = &sB[buf][wn * 64 + nt * 16 + lr][hw ? 8 : 0];
    const v8bf q0 = *(const v8bf*)p;
    const v8bf q1 = *(const v8bf*)(p + 16);
    return __builtin_shufflevector(q0, q1, 0, 1, 2, 3, 4, 5, 6, 7,
                                   8, 9, 10, 11, 12, 13, 14, 15);
  };

  v8f acc[2][4];
#pragma unroll
  for (int mt = 0; mt < 2; ++mt)
#pragma unroll
    for (int nt = 0; nt < 4; ++nt)
#pragma unroll
      for (int e = 0; e < 8; ++e) acc[mt][nt][e] = 0.0f;

  globalLoad(0);
  ldsStore(0);
  __syncthreads();

  for (int kt = 0; kt < NKI; ++kt) {
    const int buf = kt & 1;
    const bool more = (kt + 1) < NKI;
    if (more) globalLoad(kt + 1);       // overlap next HBM/L2 with WMMAs

    v16bf aH[2];
#pragma unroll
    for (int mt = 0; mt < 2; ++mt) aH[mt] = fragA(buf, mt);
#pragma unroll
    for (int nt = 0; nt < 4; ++nt) {
      const v16bf bH = fragB(buf, nt);
#pragma unroll
      for (int mt = 0; mt < 2; ++mt) {
        acc[mt][nt] = __builtin_amdgcn_wmma_f32_16x16x32_bf16(
            false, aH[mt], false, bH, (short)0, acc[mt][nt], false, false);
      }
    }

    if (more) ldsStore(buf ^ 1);        // stage kt+1 into the other buffer
    __syncthreads();
  }

  // Fused epilogue: C-layout -> VGPR r, lane<16: M=r, N=lane; lane>=16: M=r+8
  const float* xsq = sq;
  const float* csq = sq + kB;

  // xsq rows for this wave: 8 consecutive floats per mt, 8-aligned -> 2x b128
  float xs[2][8];
#pragma unroll
  for (int mt = 0; mt < 2; ++mt) {
    const int gmBase = m0 + wm * 32 + mt * 16 + hw * 8;
    const float4 a = *(const float4*)&xsq[gmBase];
    const float4 b = *(const float4*)&xsq[gmBase + 4];
    xs[mt][0] = a.x; xs[mt][1] = a.y; xs[mt][2] = a.z; xs[mt][3] = a.w;
    xs[mt][4] = b.x; xs[mt][5] = b.y; xs[mt][6] = b.z; xs[mt][7] = b.w;
  }

#pragma unroll
  for (int nt = 0; nt < 4; ++nt) {
    const int gn = n0 + wn * 64 + nt * 16 + lr;
    const float cs = csq[gn];
#pragma unroll
    for (int mt = 0; mt < 2; ++mt) {
      const int gmBase = m0 + wm * 32 + mt * 16 + hw * 8;
      float* po = out + (size_t)gmBase * kK + gn;   // one addr per (mt,nt)
#pragma unroll
      for (int r = 0; r < 8; ++r) {
        const float d2 = xs[mt][r] + cs - 2.0f * acc[mt][nt][r];
        po[r * kK] = sqrtf(fmaxf(d2, kEps));        // r*1KB -> imm offset
      }
    }
  }
}

// ---------------------------------------------------------------------------
extern "C" void kernel_launch(void* const* d_in, const int* in_sizes, int n_in,
                              void* d_out, int out_size, void* d_ws, size_t ws_size,
                              hipStream_t stream) {
  (void)in_sizes; (void)n_in; (void)out_size; (void)ws_size;
  const float* x  = (const float*)d_in[0];   // [8192, 3072] f32
  const float* c  = (const float*)d_in[1];   // [256, 3072]  f32
  float* out = (float*)d_out;                // [8192, 256]  f32
  float* ws  = (float*)d_ws;                 // [8192 + 256] f32 row norms

  rbf_sqnorm<<<kB + kK, 256, 0, stream>>>(x, c, ws);

  dim3 grid(kK / BN, kB / BM);               // (2, 64)
  rbf_gemm_wmma<<<grid, 256, 0, stream>>>(x, c, ws, out);
}
